// AttentionBsl_1829656068610
// MI455X (gfx1250) — matrix operand order repaired and checked by reading, not compiled, not run on hardware
//
#include <hip/hip_runtime.h>
#include <stdint.h>

typedef _Float16 half_t;
typedef __attribute__((ext_vector_type(16))) _Float16 v16h;
typedef __attribute__((ext_vector_type(8)))  _Float16 v8h;
typedef __attribute__((ext_vector_type(8)))  float    v8f;
typedef __attribute__((ext_vector_type(4)))  unsigned int u32x4;
typedef __attribute__((ext_vector_type(8)))  int          i32x8;
typedef __attribute__((ext_vector_type(4)))  int          i32x4;

#define DIMX   1024
#define SEQ    2048
#define BATCH  2
#define NCOL   (SEQ*BATCH)   // 4096
#define HEADS  16
#define DHEAD  64
#define INNER  1024
#define QKV_ROWS 3072

union V16 { v16h v; v8h h[2]; };

__device__ __forceinline__ v8f zero8() {
    v8f z = {0.f,0.f,0.f,0.f,0.f,0.f,0.f,0.f};
    return z;
}

// A fragment (16x32 f16): lane l (hi=l>>4) holds row m:
//   elems[0..7]  = A[m][kk + hi*8 .. +7]
//   elems[8..15] = A[m][kk + 16 + hi*8 .. +7]
__device__ __forceinline__ v16h load_frag_a(const half_t* s, int ld, int m, int kk, int lane) {
    const int hi = lane >> 4;
    const half_t* p = s + (size_t)m * ld;
    V16 u;
    u.h[0] = *(const v8h*)(p + kk + hi * 8);
    u.h[1] = *(const v8h*)(p + kk + 16 + hi * 8);
    return u.v;
}

// B fragment (32x16 f16) from LDS stored as Bs[ncol][k] (k contiguous):
//   lane l: col = given, elems[0..7] = B[k = kk + hi*8 .. +7][col], elems[8..15] = B[k = kk + 16 + hi*8 .. +7][col]
__device__ __forceinline__ v16h load_frag_b(const half_t* s, int ld, int ncol, int kk, int lane) {
    const int hi = lane >> 4;
    const half_t* p = s + (size_t)ncol * ld + kk + hi * 8;
    V16 u;
    u.h[0] = *(const v8h*)(p);
    u.h[1] = *(const v8h*)(p + 16);
    return u.v;
}

// ---------------------------------------------------------------------------
// TDM: DMA a 2D tile (tile_dim0 = 32 halves along K, tile_dim1 = 128 rows)
// from global to LDS, with hardware padding of 4 DWORDs every 16 DWORDs so
// the LDS row pitch becomes 40 halves (bank-conflict-free fragment reads).
// D# packing per CDNA5 ISA section 8 (group0: count/lds/global/type;
// group1: data_size, dims, tile dims, stride, pad fields).
// ---------------------------------------------------------------------------
__device__ __forceinline__ i32x8 tdm_make_g1(int K, int M) {
    // [15:0] wg_mask=0 | [17:16] data_size=1 (2B) | [20] pad_en=1
    // [24:22] pad_interval=3 (16 DWORDs) | [31:25] pad_amount=3 (4 DWORDs)
    i32x8 g1;
    g1[0] = (int)((1u << 16) | (1u << 20) | (3u << 22) | (3u << 25));
    g1[1] = (int)(((unsigned)K & 0xFFFFu) << 16);            // tensor_dim0 lo16
    g1[2] = (int)((((unsigned)K >> 16) & 0xFFFFu) |
                  (((unsigned)M & 0xFFFFu) << 16));          // dim0 hi | dim1 lo
    g1[3] = (int)((((unsigned)M >> 16) & 0xFFFFu) |
                  (32u << 16));                              // dim1 hi | tile_dim0=32
    g1[4] = 128;                                             // tile_dim1=128, tile_dim2=0
    g1[5] = K;                                               // tensor_dim0_stride lo32
    g1[6] = 0;                                               // stride hi | dim1_stride lo
    g1[7] = 0;
    return g1;
}

__device__ __forceinline__ void tdm_load_tile(const half_t* gsrc, unsigned lds_off,
                                              i32x8 g1) {
    unsigned long long ga = (unsigned long long)(uintptr_t)gsrc;
    u32x4 g0;
    g0[0] = 1u;                                   // count=1, user mode, no gather
    g0[1] = lds_off;                              // LDS byte address
    g0[2] = (unsigned)ga;                         // global addr [31:0]
    g0[3] = (unsigned)(ga >> 32) | (2u << 30);    // addr [56:32] | type=2
    i32x4 z4 = {0, 0, 0, 0};
#if defined(__clang_major__) && (__clang_major__ >= 23)
    i32x8 z8 = {0, 0, 0, 0, 0, 0, 0, 0};
    __builtin_amdgcn_tensor_load_to_lds(g0, g1, z4, z4, z8, 0);
#else
    __builtin_amdgcn_tensor_load_to_lds(g0, g1, z4, z4, 0);
#endif
}

// ---------------------------------------------------------------------------
// f32 -> f16 elementwise (weight conversion)
// ---------------------------------------------------------------------------
__global__ void f32_to_f16(const float* __restrict__ in, half_t* __restrict__ out, int n) {
    int i = blockIdx.x * blockDim.x + threadIdx.x;
    int stride = gridDim.x * blockDim.x;
    for (; i < n; i += stride) out[i] = (half_t)in[i];
}

// ---------------------------------------------------------------------------
// LayerNorm over leading dim axis; one thread per column c = n*BATCH + b
// ---------------------------------------------------------------------------
__global__ __launch_bounds__(64)
void ln_kernel(const float* __restrict__ x, const float* __restrict__ gamma,
               const float* __restrict__ beta, half_t* __restrict__ h) {
    const int c = blockIdx.x * 64 + threadIdx.x;
    float sum = 0.f, sq = 0.f;
    for (int d = 0; d < DIMX; d++) {
        float v = x[(size_t)d * NCOL + c];
        sum += v; sq += v * v;
    }
    const float mean = sum * (1.0f / DIMX);
    const float var  = sq * (1.0f / DIMX) - mean * mean;
    const float rstd = rsqrtf(var + 1e-5f);
    for (int d = 0; d < DIMX; d++) {
        float v = x[(size_t)d * NCOL + c];
        float y = (v - mean) * rstd * gamma[d] + beta[d];
        h[(size_t)d * NCOL + c] = (half_t)y;
    }
}

// ---------------------------------------------------------------------------
// f16 GEMM: C[M x N] = A[M x K] * B[K x N], row-major.
// 256 threads = 8 waves; block tile 128x128, BK=32; each wave computes 64x32.
// A tile DMA'd by the Tensor Data Mover (wave 0, s_wait_tensorcnt), with
// hardware-inserted padding to a 40-half LDS pitch. B tile is vector-loaded
// and transposed into LDS by all threads.
// ---------------------------------------------------------------------------
template <typename OutT>
__global__ __launch_bounds__(256)
void gemm_f16(const half_t* __restrict__ A, const half_t* __restrict__ B,
              OutT* __restrict__ C, int M, int N, int K) {
    __shared__ half_t As[128][40];   // [m][k], pitch 40 (TDM pads 4 DW / 16 DW)
    __shared__ half_t Bs[128][40];   // [n][k] (B transposed in LDS)

    const int tid  = threadIdx.x;
    const int lane = tid & 31;
    const int wave = tid >> 5;
    const int ln   = lane & 15;
    const int hi   = lane >> 4;
    const int wm   = (wave >> 2) * 64;   // 0 or 64
    const int wn   = (wave & 3) * 32;    // 0,32,64,96
    const int mBase = blockIdx.x * 128;
    const int nBase = blockIdx.y * 128;

    const i32x8 g1 = tdm_make_g1(K, M);
    const unsigned asLds = (unsigned)(uintptr_t)&As[0][0];

    v8f acc[4][2];
    #pragma unroll
    for (int i = 0; i < 4; i++)
        #pragma unroll
        for (int j = 0; j < 2; j++) acc[i][j] = zero8();

    for (int k0 = 0; k0 < K; k0 += 32) {
        __syncthreads();   // previous iteration's fragment reads complete
        if (wave == 0) {
            tdm_load_tile(&A[(size_t)mBase * K + k0], asLds, g1);
            __builtin_amdgcn_s_wait_tensorcnt(0);
        }
        // B tile (transpose): Bs[n][k] = B[(k0+k)*N + nBase+n]
        {
            int k  = tid & 31;          // 0..31
            int n0 = (tid >> 5) * 16;   // 0,16,...,112
            const half_t* src = &B[(size_t)(k0 + k) * N + nBase + n0];
            v8h t0 = *(const v8h*)(src);
            v8h t1 = *(const v8h*)(src + 8);
            #pragma unroll
            for (int j = 0; j < 8; j++) {
                Bs[n0 + j][k]     = t0[j];
                Bs[n0 + 8 + j][k] = t1[j];
            }
            if (k0 + 32 < K)
                __builtin_prefetch(&B[(size_t)(k0 + 32 + k) * N + nBase + n0], 0, 3);
        }
        __syncthreads();

        v16h af[4], bf[2];
        #pragma unroll
        for (int sm = 0; sm < 4; sm++)
            af[sm] = load_frag_a(&As[0][0], 40, wm + sm * 16 + ln, 0, lane);
        #pragma unroll
        for (int sn = 0; sn < 2; sn++)
            bf[sn] = load_frag_b(&Bs[0][0], 40, wn + sn * 16 + ln, 0, lane);
        #pragma unroll
        for (int sm = 0; sm < 4; sm++)
            #pragma unroll
            for (int sn = 0; sn < 2; sn++)
                acc[sm][sn] = __builtin_amdgcn_wmma_f32_16x16x32_f16(
                    false, af[sm], false, bf[sn], (short)0, acc[sm][sn], false, false);
    }

    #pragma unroll
    for (int sm = 0; sm < 4; sm++)
        #pragma unroll
        for (int sn = 0; sn < 2; sn++) {
            int col = nBase + wn + sn * 16 + ln;
            #pragma unroll
            for (int r = 0; r < 8; r++) {
                int row = mBase + wm + sm * 16 + hi * 8 + r;
                C[(size_t)row * N + col] = static_cast<OutT>(acc[sm][sn][r]);
            }
        }
}

// ---------------------------------------------------------------------------
// Flash-style attention. Grid: (SEQ/64, HEADS, BATCH), 128 threads = 4 waves.
// Each wave owns a 16-query strip of the 64-query tile; loops over 64-key
// tiles with online softmax. qkv is f16 [3072][4096] (col = n*BATCH + b).
// Output att is f16 [(h*64+d)][4096].
// ---------------------------------------------------------------------------
__global__ __launch_bounds__(128)
void attn_kernel(const half_t* __restrict__ qkv, half_t* __restrict__ att) {
    __shared__ half_t Qs[64][72];   // [query][d]
    __shared__ half_t Ks[64][72];   // [key][d]
    __shared__ half_t Vs[64][72];   // [d][key]  (naturally transposed for P*V)
    __shared__ half_t Ps[64][72];   // exp(S) tile, [query][key]
    __shared__ float  Ss[64][65];   // raw scores
    __shared__ float  corr_s[64];
    __shared__ float  lsum_s[64];

    const int tid  = threadIdx.x;
    const int lane = tid & 31;
    const int wave = tid >> 5;
    const int ln   = lane & 15;
    const int hi   = lane >> 4;
    const int qBase = blockIdx.x * 64;
    const int head  = blockIdx.y;
    const int b     = blockIdx.z;
    const size_t qRow0 = (size_t)(head * DHEAD) * NCOL;
    const size_t kRow0 = (size_t)(INNER + head * DHEAD) * NCOL;
    const size_t vRow0 = (size_t)(2 * INNER + head * DHEAD) * NCOL;

    // Load Q tile (coalesced over queries; transpose into LDS)
    {
        int m  = tid & 63;
        int d0 = tid >> 6;          // 0 or 1
        for (int it = 0; it < 32; it++) {
            int d = d0 + it * 2;
            Qs[m][d] = qkv[qRow0 + (size_t)d * NCOL + (size_t)(qBase + m) * BATCH + b];
        }
    }

    v8f oacc[4];
    for (int t = 0; t < 4; t++) oacc[t] = zero8();
    float m_run = -1e30f, l_run = 0.f;    // row stats, live in lanes 0..15

    for (int kt = 0; kt < SEQ / 64; kt++) {
        const int kBase = kt * 64;
        __syncthreads();
        {
            int n  = tid & 63;
            int d0 = tid >> 6;
            for (int it = 0; it < 32; it++) {
                int d = d0 + it * 2;
                Ks[n][d] = qkv[kRow0 + (size_t)d * NCOL + (size_t)(kBase + n) * BATCH + b];
                Vs[d][n] = qkv[vRow0 + (size_t)d * NCOL + (size_t)(kBase + n) * BATCH + b];
            }
        }
        __syncthreads();

        // S = Q * K^T for this wave's 16-query strip (16 x 64)
        v8f sacc[4];
        for (int t = 0; t < 4; t++) sacc[t] = zero8();
        #pragma unroll
        for (int kk = 0; kk < 64; kk += 32) {
            v16h a = load_frag_a(&Qs[0][0], 72, wave * 16 + ln, kk, lane);
            #pragma unroll
            for (int t = 0; t < 4; t++) {
                v16h bf = load_frag_b(&Ks[0][0], 72, t * 16 + ln, kk, lane);
                sacc[t] = __builtin_amdgcn_wmma_f32_16x16x32_f16(false, a, false, bf, (short)0, sacc[t], false, false);
            }
        }
        #pragma unroll
        for (int t = 0; t < 4; t++)
            #pragma unroll
            for (int r = 0; r < 8; r++)
                Ss[wave * 16 + hi * 8 + r][t * 16 + ln] = sacc[t][r];
        __syncthreads();

        // Online softmax: lanes 0..15 each own one row of the strip
        if (lane < 16) {
            const int row = wave * 16 + lane;
            const float scale = 0.125f;   // 1/sqrt(64)
            float mloc = -1e30f;
            for (int c = 0; c < 64; c++) mloc = fmaxf(mloc, Ss[row][c] * scale);
            float mnew = fmaxf(m_run, mloc);
            float corr = __expf(m_run - mnew);
            float acc = 0.f;
            for (int c = 0; c < 64; c++) {
                float e = __expf(Ss[row][c] * scale - mnew);
                acc += e;
                Ps[row][c] = (half_t)e;
            }
            l_run = l_run * corr + acc;
            m_run = mnew;
            corr_s[row] = corr;
        }
        __syncthreads();

        // Rescale running O and accumulate P * V
        #pragma unroll
        for (int r = 0; r < 8; r++) {
            float cf = corr_s[wave * 16 + hi * 8 + r];
            #pragma unroll
            for (int t = 0; t < 4; t++) oacc[t][r] *= cf;
        }
        #pragma unroll
        for (int kk = 0; kk < 64; kk += 32) {
            v16h a = load_frag_a(&Ps[0][0], 72, wave * 16 + ln, kk, lane);
            #pragma unroll
            for (int t = 0; t < 4; t++) {
                v16h bf = load_frag_b(&Vs[0][0], 72, t * 16 + ln, kk, lane);
                oacc[t] = __builtin_amdgcn_wmma_f32_16x16x32_f16(false, a, false, bf, (short)0, oacc[t], false, false);
            }
        }
    }

    if (lane < 16) lsum_s[wave * 16 + lane] = l_run;
    __syncthreads();

    // Write attention output in (h d) n b layout, f16
    #pragma unroll
    for (int r = 0; r < 8; r++) {
        int m = wave * 16 + hi * 8 + r;
        float inv = 1.0f / lsum_s[m];
        #pragma unroll
        for (int t = 0; t < 4; t++) {
            int j = t * 16 + ln;
            att[(size_t)(head * DHEAD + j) * NCOL + (size_t)(qBase + m) * BATCH + b] =
                (half_t)(oacc[t][r] * inv);
        }
    }
}

// ---------------------------------------------------------------------------
extern "C" void kernel_launch(void* const* d_in, const int* in_sizes, int n_in,
                              void* d_out, int out_size, void* d_ws, size_t ws_size,
                              hipStream_t stream) {
    const float* x      = (const float*)d_in[0];
    const float* w_qkv  = (const float*)d_in[1];
    const float* w_out  = (const float*)d_in[2];
    const float* gamma  = (const float*)d_in[3];
    const float* beta   = (const float*)d_in[4];
    float* out = (float*)d_out;

    char* ws = (char*)d_ws;
    half_t* wqkv_h = (half_t*)(ws);                 //  6 MB : w_qkv f16
    half_t* wout_h = (half_t*)(ws + 6291456);       //  2 MB : w_out f16
    half_t* h_h    = (half_t*)(ws + 8388608);       //  8 MB : LN output f16
    half_t* qkv_h  = (half_t*)(ws + 16777216);      // 24 MB : QKV f16
    half_t* att_h  = (half_t*)(ws + 41943040);      //  8 MB : attention out f16

    f32_to_f16<<<dim3(1024), dim3(256), 0, stream>>>(w_qkv, wqkv_h, QKV_ROWS * DIMX);
    f32_to_f16<<<dim3(512),  dim3(256), 0, stream>>>(w_out, wout_h, DIMX * INNER);

    ln_kernel<<<dim3(NCOL / 64), dim3(64), 0, stream>>>(x, gamma, beta, h_h);

    gemm_f16<half_t><<<dim3(QKV_ROWS / 128, NCOL / 128), dim3(256), 0, stream>>>(
        wqkv_h, h_h, qkv_h, QKV_ROWS, NCOL, DIMX);

    attn_kernel<<<dim3(SEQ / 64, HEADS, BATCH), dim3(128), 0, stream>>>(qkv_h, att_h);

    gemm_f16<float><<<dim3(DIMX / 128, NCOL / 128), dim3(256), 0, stream>>>(
        wout_h, att_h, out, DIMX, NCOL, INNER);
}
